// FastConformerBlock_89962384982156
// MI455X (gfx1250) — compile-verified, matrix operand order checked by reading, not run on hardware
//
#include <hip/hip_runtime.h>
#include <hip/hip_bf16.h>

typedef __bf16 bf16;
typedef __attribute__((ext_vector_type(16))) __bf16 v16bf;
typedef __attribute__((ext_vector_type(8)))  float  v8f;

static constexpr int   Dm   = 512;
static constexpr int   Hh   = 8;
static constexpr int   Bb   = 4;
static constexpr int   Ss   = 2048;
static constexpr int   NT   = Bb * Ss;    // 8192 token rows
static constexpr int   FH   = 4 * Dm;     // 2048
static constexpr int   KC   = 15;
static constexpr int   LEFTW = 256;
static constexpr float EPS  = 1e-5f;

#define DEV __device__ __forceinline__

DEV int kbase_of(int j, int half) { return ((j < 4) ? 0 : 16) + half * 8 + (j & 3) * 2; }

DEV v8f wmma_bf16(v16bf a, v16bf b, v8f c) {
  return __builtin_amdgcn_wmma_f32_16x16x32_bf16(false, a, false, b, (short)0, c, false, false);
}

// 16x32 bf16 fragment (A or B^T layout): per-lane contiguous K-pairs;
// lowers to two global_load_b128 at byte offsets {0,32} (half 0) / {16,48} (half 1).
DEV v16bf load_frag(const bf16* __restrict__ p, int half) {
  v16bf f;
#pragma unroll
  for (int j = 0; j < 8; ++j) {
    int kb = kbase_of(j, half);
    f[2 * j]     = p[kb];
    f[2 * j + 1] = p[kb + 1];
  }
  return f;
}

DEV float silu(float x) { return x / (1.f + __expf(-x)); }

// ---------------------------------------------------------------------------
// fp32 W[K,N] -> bf16 Wt[N,K] (transposed so B-fragment K-pairs are contiguous)
// ---------------------------------------------------------------------------
__global__ void wcvt_kernel(const float* __restrict__ W, bf16* __restrict__ Wt, int K, int N) {
  int idx = blockIdx.x * 256 + threadIdx.x;
  if (idx >= N * K) return;
  int n = idx / K, k = idx - n * K;
  Wt[idx] = (bf16)W[(size_t)k * N + n];
}

// ---------------------------------------------------------------------------
// LayerNorm: one wave32 per row of D=512
// ---------------------------------------------------------------------------
__global__ void __launch_bounds__(32)
ln_kernel(const float* __restrict__ X, const float* __restrict__ g, const float* __restrict__ b,
          bf16* __restrict__ outB, float* __restrict__ outF) {
  int row = blockIdx.x;
  int lane = threadIdx.x;
  const float* x = X + (size_t)row * Dm;
  float s = 0.f;
  for (int i = lane; i < Dm; i += 32) s += x[i];
  for (int o = 16; o; o >>= 1) s += __shfl_xor(s, o, 32);
  float mean = s * (1.f / Dm);
  float v = 0.f;
  for (int i = lane; i < Dm; i += 32) { float d = x[i] - mean; v += d * d; }
  for (int o = 16; o; o >>= 1) v += __shfl_xor(v, o, 32);
  float inv = rsqrtf(v * (1.f / Dm) + EPS);
  for (int i = lane; i < Dm; i += 32) {
    float y = (x[i] - mean) * inv * g[i] + b[i];
    if (outB) outB[(size_t)row * Dm + i] = (bf16)y;
    if (outF) outF[(size_t)row * Dm + i] = y;
  }
}

// ---------------------------------------------------------------------------
// Generic bf16 WMMA GEMM: C = act(A[M,K] @ W[K,N] + bias) [+ residual].
// Wave tile = 64(M) x 64(N): 4 A-frags x 4 B-frags -> 16 WMMAs per 32-wide K
// step (~32 flop/byte from L2). K unrolled by 64 with ping-pong fragment
// double-buffering so each 16-WMMA burst overlaps the next step's loads.
// Compile-time ACT / residual / output-format flags keep the epilogue
// branch-free. Requires K % 64 == 0 (true for K = 512, 2048).
// ---------------------------------------------------------------------------
template <int ACT, bool HAS_RES, bool W_F, bool W_B>
__global__ void __launch_bounds__(128)
gemm_bf16_kernel(const bf16* __restrict__ A, const bf16* __restrict__ Wt,
                 const float* __restrict__ bias, const float* __restrict__ residual,
                 float res_scale, float* __restrict__ outF, bf16* __restrict__ outB,
                 int M, int N, int K) {
  int lane = threadIdx.x & 31;
  int wave = threadIdx.x >> 5;
  int tile = blockIdx.x * 4 + wave;
  int tilesN = N >> 6;
  int tm = tile / tilesN, tn = tile - tm * tilesN;
  if (tm * 64 >= M) return;
  int m = lane & 15, half = lane >> 4;

  const bf16* arow[4];
  const bf16* wrow[4];
#pragma unroll
  for (int mt = 0; mt < 4; ++mt) arow[mt] = A + (size_t)(tm * 64 + mt * 16 + m) * K;
#pragma unroll
  for (int nt = 0; nt < 4; ++nt) wrow[nt] = Wt + (size_t)(tn * 64 + nt * 16 + m) * K;

  v8f acc[4][4];
#pragma unroll
  for (int mt = 0; mt < 4; ++mt)
#pragma unroll
    for (int nt = 0; nt < 4; ++nt) acc[mt][nt] = v8f{};

  v16bf a0[4], b0[4], a1[4], b1[4];
#pragma unroll
  for (int mt = 0; mt < 4; ++mt) a0[mt] = load_frag(arow[mt], half);
#pragma unroll
  for (int nt = 0; nt < 4; ++nt) b0[nt] = load_frag(wrow[nt], half);

  for (int kk = 0; kk < K; kk += 64) {
    // issue loads for kk+32 (always in range: K % 64 == 0)
#pragma unroll
    for (int mt = 0; mt < 4; ++mt) a1[mt] = load_frag(arow[mt] + kk + 32, half);
#pragma unroll
    for (int nt = 0; nt < 4; ++nt) b1[nt] = load_frag(wrow[nt] + kk + 32, half);
    // compute on buffer 0 while buffer-1 loads are in flight
#pragma unroll
    for (int nt = 0; nt < 4; ++nt)
#pragma unroll
      for (int mt = 0; mt < 4; ++mt)
        acc[mt][nt] = wmma_bf16(a0[mt], b0[nt], acc[mt][nt]);
    // issue loads for kk+64
    if (kk + 64 < K) {
#pragma unroll
      for (int mt = 0; mt < 4; ++mt) a0[mt] = load_frag(arow[mt] + kk + 64, half);
#pragma unroll
      for (int nt = 0; nt < 4; ++nt) b0[nt] = load_frag(wrow[nt] + kk + 64, half);
    }
    // compute on buffer 1 while buffer-0 loads are in flight
#pragma unroll
    for (int nt = 0; nt < 4; ++nt)
#pragma unroll
      for (int mt = 0; mt < 4; ++mt)
        acc[mt][nt] = wmma_bf16(a1[mt], b1[nt], acc[mt][nt]);
  }

#pragma unroll
  for (int mt = 0; mt < 4; ++mt)
#pragma unroll
    for (int nt = 0; nt < 4; ++nt) {
      int col = tn * 64 + nt * 16 + m;
      float bv = bias[col];
#pragma unroll
      for (int r = 0; r < 8; ++r) {
        int row = tm * 64 + mt * 16 + r + half * 8;
        float v = acc[mt][nt][r] + bv;
        if (ACT == 1) v = silu(v);
        size_t idx = (size_t)row * N + col;
        float o = HAS_RES ? (residual[idx] + res_scale * v) : v;
        if (W_F) outF[idx] = o;
        if (W_B) outB[idx] = (bf16)o;
      }
    }
}

// ---------------------------------------------------------------------------
// Banded flash attention: 1 wave per (b, h, 16-query tile).
// qkv: bf16 [NT, 3D] (q|k|v). out: bf16 [NT, D] (heads interleaved as reshape).
// ---------------------------------------------------------------------------
__global__ void __launch_bounds__(32)
attn_kernel(const bf16* __restrict__ qkv, bf16* __restrict__ outb) {
  __shared__ bf16 pl[16 * 32];   // P tile staging (C-layout -> A-layout repack)
  const int QT = Ss / 16;
  int lane = threadIdx.x;
  int qt = blockIdx.x % QT;
  int h  = (blockIdx.x / QT) % Hh;
  int b  = blockIdx.x / (QT * Hh);
  int q0 = qt * 16;
  int m = lane & 15, half = lane >> 4;
  const float scale = 0.125f;  // (D/H)^-0.5 = 64^-0.5
  const int W3 = 3 * Dm;

  // Q as two 16x32 A-fragments (dh = 64)
  v16bf aq[2];
  {
    const bf16* qrow = qkv + (size_t)(b * Ss + q0 + m) * W3 + h * 64;
    aq[0] = load_frag(qrow, half);
    aq[1] = load_frag(qrow + 32, half);
  }

  float mrow[8], lrow[8];
  v8f o[4] = {v8f{}, v8f{}, v8f{}, v8f{}};
#pragma unroll
  for (int r = 0; r < 8; ++r) { mrow[r] = -1e30f; lrow[r] = 0.f; }

  int kb0 = (q0 > LEFTW) ? ((q0 - LEFTW) & ~31) : 0;
  for (int kb = kb0; kb <= q0 + 15; kb += 32) {
    // S = Q @ K^T for 32 keys (two 16-key chunks)
    v8f sc[2] = {v8f{}, v8f{}};
#pragma unroll
    for (int c = 0; c < 2; ++c) {
      int j = kb + c * 16 + m;
      int jc = (j < Ss) ? j : (Ss - 1);
      const bf16* krow = qkv + (size_t)(b * Ss + jc) * W3 + Dm + h * 64;
      v16bf bk0 = load_frag(krow, half);
      v16bf bk1 = load_frag(krow + 32, half);
      sc[c] = wmma_bf16(aq[0], bk0, sc[c]);
      sc[c] = wmma_bf16(aq[1], bk1, sc[c]);
    }
    // Online softmax over this 32-key block (band mask j in [i-256, i])
#pragma unroll
    for (int r = 0; r < 8; ++r) {
      int i  = q0 + r + half * 8;
      int j0 = kb + m, j1 = j0 + 16;
      bool ok0 = (j0 <= i) && (j0 + LEFTW >= i);
      bool ok1 = (j1 <= i) && (j1 + LEFTW >= i);
      float s0 = ok0 ? sc[0][r] * scale : -1e30f;
      float s1 = ok1 ? sc[1][r] * scale : -1e30f;
      float mx = fmaxf(s0, s1);
      for (int off = 8; off; off >>= 1) mx = fmaxf(mx, __shfl_xor(mx, off, 32));
      float mnew = fmaxf(mrow[r], mx);
      float corr = __expf(mrow[r] - mnew);
      float p0 = ok0 ? __expf(s0 - mnew) : 0.f;
      float p1 = ok1 ? __expf(s1 - mnew) : 0.f;
      float ps = p0 + p1;
      for (int off = 8; off; off >>= 1) ps += __shfl_xor(ps, off, 32);
      lrow[r] = lrow[r] * corr + ps;
      mrow[r] = mnew;
      o[0][r] *= corr; o[1][r] *= corr; o[2][r] *= corr; o[3][r] *= corr;
      pl[(r + half * 8) * 32 + m]      = (bf16)p0;
      pl[(r + half * 8) * 32 + 16 + m] = (bf16)p1;
    }
    __syncthreads();
    // P (16x32) repacked to A-fragment, then O += P @ V
    v16bf ap;
#pragma unroll
    for (int j = 0; j < 8; ++j) {
      int kbb = kbase_of(j, half);
      ap[2 * j]     = pl[m * 32 + kbb];
      ap[2 * j + 1] = pl[m * 32 + kbb + 1];
    }
#pragma unroll
    for (int nt = 0; nt < 4; ++nt) {
      int col = nt * 16 + m;
      v16bf bv;
#pragma unroll
      for (int j = 0; j < 8; ++j) {
        int kbb = kbase_of(j, half);
        int ja = kb + kbb, jb2 = ja + 1;
        int jca = (ja < Ss) ? ja : (Ss - 1);
        int jcb = (jb2 < Ss) ? jb2 : (Ss - 1);
        bv[2 * j]     = qkv[(size_t)(b * Ss + jca) * W3 + 2 * Dm + h * 64 + col];
        bv[2 * j + 1] = qkv[(size_t)(b * Ss + jcb) * W3 + 2 * Dm + h * 64 + col];
      }
      o[nt] = wmma_bf16(ap, bv, o[nt]);
    }
    __syncthreads();
  }

#pragma unroll
  for (int nt = 0; nt < 4; ++nt)
#pragma unroll
    for (int r = 0; r < 8; ++r) {
      int row = q0 + r + half * 8;
      float v = o[nt][r] / lrow[r];
      outb[(size_t)(b * Ss + row) * Dm + h * 64 + nt * 16 + m] = (bf16)v;
    }
}

// ---------------------------------------------------------------------------
// GLU over channels: [T, 2D] -> [T, D]
// ---------------------------------------------------------------------------
__global__ void glu_kernel(const bf16* __restrict__ in, bf16* __restrict__ out) {
  int idx = blockIdx.x * 256 + threadIdx.x;
  if (idx >= NT * Dm) return;
  int t = idx / Dm, d = idx - t * Dm;
  float a = (float)in[(size_t)t * (2 * Dm) + d];
  float g = (float)in[(size_t)t * (2 * Dm) + Dm + d];
  out[idx] = (bf16)(a * (1.f / (1.f + __expf(-g))));
}

// ---------------------------------------------------------------------------
// Causal depthwise conv (K=15) + eval BN + SiLU
// ---------------------------------------------------------------------------
__global__ void dwconv_kernel(const bf16* __restrict__ in, const float* __restrict__ w,
                              const float* __restrict__ wb, const float* __restrict__ bng,
                              const float* __restrict__ bnb, const float* __restrict__ rm,
                              const float* __restrict__ rv, bf16* __restrict__ out) {
  int idx = blockIdx.x * 256 + threadIdx.x;
  if (idx >= NT * Dm) return;
  int t = idx / Dm, d = idx - t * Dm;
  int b = t / Ss, s = t - b * Ss;
  float acc = 0.f;
#pragma unroll
  for (int k = 0; k < KC; ++k) {
    int si = s - (KC - 1) + k;
    if (si >= 0) acc += (float)in[(size_t)(b * Ss + si) * Dm + d] * w[k * Dm + d];
  }
  acc += wb[d];
  acc = (acc - rm[d]) * rsqrtf(rv[d] + EPS) * bng[d] + bnb[d];
  out[idx] = (bf16)silu(acc);
}

// ---------------------------------------------------------------------------
// Host orchestration
// ---------------------------------------------------------------------------
// Three GEMM flavors actually used:
//   silu  -> bf16 out               (FFN expand, no residual)
//   plain -> bf16 out               (qkv / pw1, no residual)
//   plain + residual -> fp32 out    (FFN contract / wo / pw2)
static inline void launch_gemm_silu_b(const bf16* A, const bf16* Wt, const float* bias,
                                      bf16* outB, int M, int N, int K, hipStream_t st) {
  int blocks = (M / 64) * (N / 64) / 4;
  gemm_bf16_kernel<1, false, false, true><<<blocks, 128, 0, st>>>(
      A, Wt, bias, nullptr, 0.f, nullptr, outB, M, N, K);
}
static inline void launch_gemm_b(const bf16* A, const bf16* Wt, const float* bias,
                                 bf16* outB, int M, int N, int K, hipStream_t st) {
  int blocks = (M / 64) * (N / 64) / 4;
  gemm_bf16_kernel<0, false, false, true><<<blocks, 128, 0, st>>>(
      A, Wt, bias, nullptr, 0.f, nullptr, outB, M, N, K);
}
static inline void launch_gemm_res_f(const bf16* A, const bf16* Wt, const float* bias,
                                     const float* res, float res_scale, float* outF,
                                     int M, int N, int K, hipStream_t st) {
  int blocks = (M / 64) * (N / 64) / 4;
  gemm_bf16_kernel<0, true, true, false><<<blocks, 128, 0, st>>>(
      A, Wt, bias, res, res_scale, outF, nullptr, M, N, K);
}

extern "C" void kernel_launch(void* const* d_in, const int* in_sizes, int n_in,
                              void* d_out, int out_size, void* d_ws, size_t ws_size,
                              hipStream_t stream) {
  const float* x      = (const float*)d_in[0];
  const float* f1n_g  = (const float*)d_in[1];
  const float* f1n_b  = (const float*)d_in[2];
  const float* f1_w1  = (const float*)d_in[3];
  const float* f1_b1  = (const float*)d_in[4];
  const float* f1_w2  = (const float*)d_in[5];
  const float* f1_b2  = (const float*)d_in[6];
  const float* an_g   = (const float*)d_in[7];
  const float* an_b   = (const float*)d_in[8];
  const float* wqkv   = (const float*)d_in[9];
  const float* bqkv   = (const float*)d_in[10];
  const float* wo     = (const float*)d_in[11];
  const float* bo     = (const float*)d_in[12];
  const float* cn_g   = (const float*)d_in[13];
  const float* cn_b   = (const float*)d_in[14];
  const float* pw1_w  = (const float*)d_in[15];
  const float* pw1_b  = (const float*)d_in[16];
  const float* dw_w   = (const float*)d_in[17];
  const float* dw_b   = (const float*)d_in[18];
  const float* bn_g   = (const float*)d_in[19];
  const float* bn_b   = (const float*)d_in[20];
  const float* bn_rm  = (const float*)d_in[21];
  const float* bn_rv  = (const float*)d_in[22];
  const float* pw2_w  = (const float*)d_in[23];
  const float* pw2_b  = (const float*)d_in[24];
  const float* f2n_g  = (const float*)d_in[25];
  const float* f2n_b  = (const float*)d_in[26];
  const float* f2_w1  = (const float*)d_in[27];
  const float* f2_b1  = (const float*)d_in[28];
  const float* f2_w2  = (const float*)d_in[29];
  const float* f2_b2  = (const float*)d_in[30];
  const float* fn_g   = (const float*)d_in[31];
  const float* fn_b   = (const float*)d_in[32];

  size_t off = 0;
  auto alloc = [&](size_t bytes) -> void* {
    void* p = (char*)d_ws + off;
    off += (bytes + 255) & ~(size_t)255;
    return p;
  };

  bf16* twF1a = (bf16*)alloc((size_t)FH * Dm * 2);      // f1_w1^T [2048,512]
  bf16* twF1b = (bf16*)alloc((size_t)Dm * FH * 2);      // f1_w2^T [512,2048]
  bf16* twQKV = (bf16*)alloc((size_t)3 * Dm * Dm * 2);  // wqkv^T  [1536,512]
  bf16* twWO  = (bf16*)alloc((size_t)Dm * Dm * 2);
  bf16* twP1  = (bf16*)alloc((size_t)2 * Dm * Dm * 2);  // pw1^T   [1024,512]
  bf16* twP2  = (bf16*)alloc((size_t)Dm * Dm * 2);
  bf16* twF2a = (bf16*)alloc((size_t)FH * Dm * 2);
  bf16* twF2b = (bf16*)alloc((size_t)Dm * FH * 2);

  bf16* lnb   = (bf16*)alloc((size_t)NT * Dm * 2);       // LN output (bf16 GEMM input)
  bf16* h1    = (bf16*)alloc((size_t)NT * FH * 2);       // FFN hidden / pw1 out
  bf16* qkvb  = (bf16*)alloc((size_t)NT * 3 * Dm * 2);
  bf16* attnb = (bf16*)alloc((size_t)NT * Dm * 2);
  bf16* glub  = (bf16*)alloc((size_t)NT * Dm * 2);
  bf16* dwob  = (bf16*)alloc((size_t)NT * Dm * 2);
  float* x1   = (float*)alloc((size_t)NT * Dm * 4);
  float* x2   = (float*)alloc((size_t)NT * Dm * 4);
  (void)ws_size; (void)in_sizes; (void)n_in; (void)out_size;

  auto cvt = [&](const float* W, bf16* Wt, int K, int N) {
    wcvt_kernel<<<(N * K + 255) / 256, 256, 0, stream>>>(W, Wt, K, N);
  };
  cvt(f1_w1, twF1a, Dm, FH);
  cvt(f1_w2, twF1b, FH, Dm);
  cvt(wqkv,  twQKV, Dm, 3 * Dm);
  cvt(wo,    twWO,  Dm, Dm);
  cvt(pw1_w, twP1,  Dm, 2 * Dm);
  cvt(pw2_w, twP2,  Dm, Dm);
  cvt(f2_w1, twF2a, Dm, FH);
  cvt(f2_w2, twF2b, FH, Dm);

  // FFN1 half-residual: x1 = x + 0.5 * ffn(ln(x))
  ln_kernel<<<NT, 32, 0, stream>>>(x, f1n_g, f1n_b, lnb, nullptr);
  launch_gemm_silu_b(lnb, twF1a, f1_b1, h1, NT, FH, Dm, stream);
  launch_gemm_res_f(h1, twF1b, f1_b2, x, 0.5f, x1, NT, Dm, FH, stream);

  // Attention: x2 = x1 + attn(ln(x1))
  ln_kernel<<<NT, 32, 0, stream>>>(x1, an_g, an_b, lnb, nullptr);
  launch_gemm_b(lnb, twQKV, bqkv, qkvb, NT, 3 * Dm, Dm, stream);
  attn_kernel<<<Bb * Hh * (Ss / 16), 32, 0, stream>>>(qkvb, attnb);
  launch_gemm_res_f(attnb, twWO, bo, x1, 1.f, x2, NT, Dm, Dm, stream);

  // Conv module: x1 (reused) = x2 + pw2(silu(bn(dwconv(glu(pw1(ln(x2)))))))
  ln_kernel<<<NT, 32, 0, stream>>>(x2, cn_g, cn_b, lnb, nullptr);
  launch_gemm_b(lnb, twP1, pw1_b, h1, NT, 2 * Dm, Dm, stream);
  glu_kernel<<<(NT * Dm + 255) / 256, 256, 0, stream>>>(h1, glub);
  dwconv_kernel<<<(NT * Dm + 255) / 256, 256, 0, stream>>>(glub, dw_w, dw_b, bn_g, bn_b,
                                                           bn_rm, bn_rv, dwob);
  launch_gemm_res_f(dwob, twP2, pw2_b, x2, 1.f, x1, NT, Dm, Dm, stream);

  // FFN2 half-residual: x2 (reused) = x1 + 0.5 * ffn(ln(x1))
  ln_kernel<<<NT, 32, 0, stream>>>(x1, f2n_g, f2n_b, lnb, nullptr);
  launch_gemm_silu_b(lnb, twF2a, f2_b1, h1, NT, FH, Dm, stream);
  launch_gemm_res_f(h1, twF2b, f2_b2, x1, 0.5f, x2, NT, Dm, FH, stream);

  // Final LN -> fp32 output
  ln_kernel<<<NT, 32, 0, stream>>>(x2, fn_g, fn_b, nullptr, (float*)d_out);
}
